// Samba_Layer_25503515803686
// MI455X (gfx1250) — matrix operand order repaired and checked by reading, not compiled, not run on hardware
//
#include <hip/hip_runtime.h>
#include <hip/hip_bf16.h>
#include <stdint.h>

// Model dims (fixed by reference)
#define LSEQ   2048
#define DMODEL 512
#define NLAYER 4
#define NSTATE 16
#define KCONV  4
#define DINNER 1024
#define RRANK  32
#define HMLP   2048

typedef __attribute__((ext_vector_type(16))) _Float16 v16h;
typedef __attribute__((ext_vector_type(8)))  float    v8f;
typedef __attribute__((ext_vector_type(4))) unsigned int u32x4_t;
typedef __attribute__((ext_vector_type(8))) int i32x8_t;
typedef __attribute__((ext_vector_type(4))) int i32x4_t;

#ifdef __has_builtin
#if __has_builtin(__builtin_amdgcn_tensor_load_to_lds)
#define HAVE_TDM 1
#endif
#endif
#ifndef HAVE_TDM
#define HAVE_TDM 0
#endif

__device__ __forceinline__ float sigmoidf_(float x){ return 1.0f/(1.0f+__expf(-x)); }
__device__ __forceinline__ float siluf_(float x){ return x*sigmoidf_(x); }
__device__ __forceinline__ float softplusf_(float x){ return (x>20.0f)? x : log1pf(__expf(x)); }
__device__ __forceinline__ float geluf_(float x){
  float x3 = x*x*x;
  return 0.5f*x*(1.0f+tanhf(0.7978845608028654f*(x+0.044715f*x3)));
}

__device__ __forceinline__ float block_reduce(float v, float* sbuf){
  int tid = threadIdx.x;
  sbuf[tid] = v; __syncthreads();
  for (int s = 128; s > 0; s >>= 1){
    if (tid < s) sbuf[tid] += sbuf[tid+s];
    __syncthreads();
  }
  float r = sbuf[0]; __syncthreads();
  return r;
}

// ---------------- LayerNorm (f32 out) ----------------
__global__ __launch_bounds__(256) void ln_kernel(const float* __restrict__ in,
                                                 const float* __restrict__ w,
                                                 const float* __restrict__ b,
                                                 float* __restrict__ out, int cols){
  __shared__ float sbuf[256];
  int row = blockIdx.x;
  const float* x = in + (size_t)row*cols;
  float s = 0.f, s2 = 0.f;
  for (int c = threadIdx.x; c < cols; c += blockDim.x){ float v = x[c]; s += v; s2 += v*v; }
  s  = block_reduce(s,  sbuf);
  s2 = block_reduce(s2, sbuf);
  float mu  = s / (float)cols;
  float var = s2 / (float)cols - mu*mu;
  float inv = rsqrtf(var + 1e-5f);
  float* o = out + (size_t)row*cols;
  for (int c = threadIdx.x; c < cols; c += blockDim.x)
    o[c] = (x[c]-mu)*inv*w[c] + b[c];
}

// ---------------- LayerNorm (f16 out, for GEMM A) ----------------
__global__ __launch_bounds__(256) void ln_f16_kernel(const float* __restrict__ in,
                                                     const float* __restrict__ w,
                                                     const float* __restrict__ b,
                                                     _Float16* __restrict__ out, int cols){
  __shared__ float sbuf[256];
  int row = blockIdx.x;
  const float* x = in + (size_t)row*cols;
  float s = 0.f, s2 = 0.f;
  for (int c = threadIdx.x; c < cols; c += blockDim.x){ float v = x[c]; s += v; s2 += v*v; }
  s  = block_reduce(s,  sbuf);
  s2 = block_reduce(s2, sbuf);
  float mu  = s / (float)cols;
  float var = s2 / (float)cols - mu*mu;
  float inv = rsqrtf(var + 1e-5f);
  _Float16* o = out + (size_t)row*cols;
  for (int c = threadIdx.x; c < cols; c += blockDim.x)
    o[c] = (_Float16)((x[c]-mu)*inv*w[c] + b[c]);
}

// ---------------- residual accumulate + RMSNorm -> f16 ----------------
__global__ __launch_bounds__(256) void residual_rms_kernel(const float* __restrict__ h,
                                                           float* __restrict__ residual,
                                                           const float* __restrict__ w,
                                                           _Float16* __restrict__ hn16,
                                                           int cols, int accumulate){
  __shared__ float sbuf[256];
  int row = blockIdx.x;
  const float* hp = h + (size_t)row*cols;
  float* rp = residual + (size_t)row*cols;
  float s2 = 0.f;
  for (int c = threadIdx.x; c < cols; c += blockDim.x){
    float v = hp[c] + (accumulate ? rp[c] : 0.0f);
    rp[c] = v;
    s2 += v*v;
  }
  s2 = block_reduce(s2, sbuf);
  float inv = rsqrtf(s2/(float)cols + 1e-5f);
  _Float16* o = hn16 + (size_t)row*cols;
  for (int c = threadIdx.x; c < cols; c += blockDim.x)
    o[c] = (_Float16)(rp[c]*inv*w[c]);
}

// ---------------- final: hout = shortcut + rms(h+residual)*w ----------------
__global__ __launch_bounds__(256) void final_rms_add_kernel(const float* __restrict__ h,
                                                            const float* __restrict__ residual,
                                                            const float* __restrict__ w,
                                                            const float* __restrict__ shortcut,
                                                            float* __restrict__ hout, int cols){
  __shared__ float sbuf[256];
  int row = blockIdx.x;
  const float* hp = h + (size_t)row*cols;
  const float* rp = residual + (size_t)row*cols;
  const float* sp = shortcut + (size_t)row*cols;
  float s2 = 0.f;
  for (int c = threadIdx.x; c < cols; c += blockDim.x){
    float v = hp[c] + rp[c];
    s2 += v*v;
  }
  s2 = block_reduce(s2, sbuf);
  float inv = rsqrtf(s2/(float)cols + 1e-5f);
  float* o = hout + (size_t)row*cols;
  for (int c = threadIdx.x; c < cols; c += blockDim.x){
    float v = hp[c] + rp[c];
    o[c] = sp[c] + v*inv*w[c];
  }
}

// ---------------- depthwise causal conv (K=4) + bias + SiLU ----------------
__global__ __launch_bounds__(256) void conv_silu_kernel(const float* __restrict__ xz,
                                                        const float* __restrict__ cw,
                                                        const float* __restrict__ cb,
                                                        float* __restrict__ u,
                                                        _Float16* __restrict__ u16){
  int idx = blockIdx.x*blockDim.x + threadIdx.x;
  if (idx >= LSEQ*DINNER) return;
  int t = idx / DINNER;
  int d = idx - t*DINNER;
  float acc = cb[d];
#pragma unroll
  for (int k = 0; k < KCONV; ++k){
    int tt = t + k - (KCONV-1);
    if (tt >= 0) acc += cw[d*KCONV + k] * xz[(size_t)tt*(2*DINNER) + d];
  }
  float s = siluf_(acc);
  u[idx] = s;
  u16[idx] = (_Float16)s;
}

// ---------------- sequential SSM scan: one thread per channel ----------------
__global__ __launch_bounds__(256) void ssm_scan_kernel(const float* __restrict__ delta,
                                                       const float* __restrict__ dbc,
                                                       const float* __restrict__ u,
                                                       const float* __restrict__ A_log,
                                                       const float* __restrict__ Dp,
                                                       float* __restrict__ y){
  int d = blockIdx.x*blockDim.x + threadIdx.x;
  if (d >= DINNER) return;
  float A[NSTATE];
#pragma unroll
  for (int n = 0; n < NSTATE; ++n) A[n] = -__expf(A_log[d*NSTATE + n]);
  float Dpd = Dp[d];
  float hs[NSTATE];
#pragma unroll
  for (int n = 0; n < NSTATE; ++n) hs[n] = 0.0f;

  for (int t = 0; t < LSEQ; ++t){
    float dt = delta[(size_t)t*DINNER + d];
    float ut = u[(size_t)t*DINNER + d];
    float du = dt*ut;
    float acc = ut*Dpd;
    const float* bc = dbc + (size_t)t*(RRANK + 2*NSTATE);
#pragma unroll
    for (int n = 0; n < NSTATE; ++n){
      float dA = __expf(dt*A[n]);
      hs[n] = dA*hs[n] + du*bc[RRANK + n];          // B_t[n]
      acc  += hs[n]*bc[RRANK + NSTATE + n];         // C_t[n]
    }
    y[(size_t)t*DINNER + d] = acc;
  }
}

// ---------------- gate: yg16 = y * silu(z) ----------------
__global__ __launch_bounds__(256) void gate_kernel(const float* __restrict__ y,
                                                   const float* __restrict__ xz,
                                                   _Float16* __restrict__ yg16){
  int idx = blockIdx.x*blockDim.x + threadIdx.x;
  if (idx >= LSEQ*DINNER) return;
  int t = idx / DINNER;
  int d = idx - t*DINNER;
  float z = xz[(size_t)t*(2*DINNER) + DINNER + d];
  yg16[idx] = (_Float16)(y[idx]*siluf_(z));
}

// ---------------- f32 -> f16 convert ----------------
__global__ __launch_bounds__(256) void cvt_f16_kernel(const float* __restrict__ in,
                                                      _Float16* __restrict__ out, int n){
  int i = blockIdx.x*blockDim.x + threadIdx.x;
  if (i < n) out[i] = (_Float16)in[i];
}

#if HAVE_TDM
// Issue one TDM descriptor: copy the 16 x Kdim f16 weight tile starting at
// `gaddr` (row stride ldb elements) into LDS at byte offset ldsOff.
// D# packing per CDNA5 ISA.
__device__ __forceinline__ void tdm_load_b_tile(uint32_t ldsOff, uint64_t gaddr,
                                                int Kdim, int Nrows, int ldb){
  u32x4_t g0;
  g0[0] = 1u;                                            // count=1, is_restore=0, gather=0
  g0[1] = ldsOff;                                        // lds_addr [63:32]
  g0[2] = (uint32_t)(gaddr & 0xffffffffull);             // global_addr [95:64]
  g0[3] = (uint32_t)((gaddr >> 32) & 0x01ffffffull)      // global_addr [120:96]
        | (2u << 30);                                    // type=2 (image) [127:126]
  uint32_t td0 = (uint32_t)Kdim;                         // tensor_dim0
  uint32_t td1 = (uint32_t)Nrows;                        // tensor_dim1
  uint32_t tl0 = (uint32_t)Kdim;                         // tile_dim0
  uint32_t tl1 = 16u;                                    // tile_dim1 (rows)
  uint64_t s0  = (uint64_t)ldb;                          // tensor_dim0_stride
  uint64_t s1  = (uint64_t)ldb * 16u;                    // tensor_dim1_stride (unused, tile_dim2=0)
  i32x8_t g1;
  g1[0] = (int)(1u << 16);                               // workgroup_mask=0, data_size=1 (2B)
  g1[1] = (int)((td0 & 0xffffu) << 16);                  // tensor_dim0 [63:48]
  g1[2] = (int)((td0 >> 16) | ((td1 & 0xffffu) << 16));  // dim0 hi [79:64], dim1 lo [95:80]
  g1[3] = (int)((td1 >> 16) | (tl0 << 16));              // dim1 hi [111:96], tile_dim0 [127:112]
  g1[4] = (int)(tl1);                                    // tile_dim1 [143:128], tile_dim2=0
  g1[5] = (int)(uint32_t)(s0 & 0xffffffffull);           // stride0 [191:160]
  g1[6] = (int)((uint32_t)(s0 >> 32) | ((uint32_t)(s1 & 0xffffull) << 16)); // [207:192],[223:208]
  g1[7] = (int)(uint32_t)(s1 >> 16);                     // stride1 [255:224]
  i32x4_t gz = {0,0,0,0};
#if __clang_major__ >= 23
  i32x8_t gz8 = {0,0,0,0,0,0,0,0};
  __builtin_amdgcn_tensor_load_to_lds(g0, g1, gz, gz, gz8, 0);
#else
  __builtin_amdgcn_tensor_load_to_lds(g0, g1, gz, gz, 0);
#endif
}
#endif

// ---------------- WMMA GEMM: C[M,N] = A[M,K] * W[N,K]^T (+epilogue) ----------------
// One block = 8 waves = 8 tile-rows x 1 tile-col. The 16 x Kdim weight tile for the
// block's tile-col is staged into LDS once (TDM DMA when available), then every wave
// reads B fragments from LDS (ds_load) while streaming its own A rows from global.
// EP: 0 = none, 1 = bias+softplus, 2 = bias+gelu, 3 = bias+add(addsrc)
template<int EP>
__global__ __launch_bounds__(256) void wmma_gemm_kernel(const _Float16* __restrict__ A, int lda,
                                                        const _Float16* __restrict__ Bw, int ldb,
                                                        float* __restrict__ C, int ldc,
                                                        const float* __restrict__ bias,
                                                        const float* __restrict__ addsrc,
                                                        int M, int N, int Kdim){
  extern __shared__ _Float16 smemB[];   // 16 x Kdim
  const int wave = threadIdx.x >> 5;
  const int lane = threadIdx.x & 31;
  const int ntile = N >> 4;
  const int tc = blockIdx.x % ntile;
  const int tr = (blockIdx.x / ntile) * 8 + wave;   // M/16 is a multiple of 8 for all calls
  const _Float16* Btile = Bw + (size_t)(tc*16)*ldb;

#if HAVE_TDM
  if (wave == 0){
    tdm_load_b_tile((uint32_t)(uintptr_t)smemB, (uint64_t)(uintptr_t)Btile, Kdim, N, ldb);
    __builtin_amdgcn_s_wait_tensorcnt(0);
  }
#else
  for (int idx = threadIdx.x; idx < 16*Kdim; idx += 256){
    int r = idx / Kdim, c0 = idx - r*Kdim;
    smemB[idx] = Btile[(size_t)r*ldb + c0];
  }
#endif
  __syncthreads();

  const int mr   = lane & 15;
  const int half = lane >> 4;
  const _Float16* Ap = A + (size_t)(tr*16 + mr)*lda;
  const _Float16* Bp = smemB + (size_t)mr*Kdim;      // lane's N-row of the LDS tile

  union Frag { v16h v; uint32_t u[8]; };
  Frag fa, fb;
  v8f acc = {0.f,0.f,0.f,0.f,0.f,0.f,0.f,0.f};

  for (int k0 = 0; k0 < Kdim; k0 += 32){
    if (k0 + 32 < Kdim) __builtin_prefetch(Ap + k0 + 32, 0, 1);
#pragma unroll
    for (int i = 0; i < 8; ++i){
      // A 16x32 f16 layout (ISA 05_wmma 7.12.2): lanes 0-15 M=lane, v0-3 K=0..7 pairs,
      // v4-7 K=16..23; lanes 16-31 get +8 on the K pair base.
      int ka = k0 + ((i>>2)<<4) + (half<<3) + ((i&3)<<1);
      fa.u[i] = *reinterpret_cast<const uint32_t*>(Ap + ka);
      // B 32x16 f16 layout: lane = N; same K order as A (v0-3 K=0..7, v4-7 K=16..23; lanes 16-31 +8)
      int kb = k0 + ((i>>2)<<4) + (half<<3) + ((i&3)<<1);
      fb.u[i] = *reinterpret_cast<const uint32_t*>(Bp + kb);
    }
    acc = __builtin_amdgcn_wmma_f32_16x16x32_f16(false, fa.v, false, fb.v,
                                                 (short)0, acc, false, false);
  }

  int col = tc*16 + mr;
  float bv = 0.0f;
  if (EP != 0) bv = bias[col];
#pragma unroll
  for (int i = 0; i < 8; ++i){
    int row = tr*16 + half*8 + i;   // C/D layout: VGPR i -> M = i + 8*half
    float v = acc[i];
    if (EP == 1)      v = softplusf_(v + bv);
    else if (EP == 2) v = geluf_(v + bv);
    else if (EP == 3) v = v + bv + addsrc[(size_t)row*ldc + col];
    C[(size_t)row*ldc + col] = v;
  }
}

// ==================================================================
extern "C" void kernel_launch(void* const* d_in, const int* in_sizes, int n_in,
                              void* d_out, int out_size, void* d_ws, size_t ws_size,
                              hipStream_t stream){
  const float* hidden      = (const float*)d_in[0];
  const float* ln1_w       = (const float*)d_in[1];
  const float* ln1_b       = (const float*)d_in[2];
  const float* in_proj_w   = (const float*)d_in[3];
  const float* conv_w      = (const float*)d_in[4];
  const float* conv_b      = (const float*)d_in[5];
  const float* x_proj_w    = (const float*)d_in[6];
  const float* dt_proj_w   = (const float*)d_in[7];
  const float* dt_proj_b   = (const float*)d_in[8];
  const float* A_log       = (const float*)d_in[9];
  const float* D_param     = (const float*)d_in[10];
  const float* out_proj_w  = (const float*)d_in[11];
  const float* block_norm_w= (const float*)d_in[12];
  const float* norm_f_w    = (const float*)d_in[13];
  const float* ln2_w       = (const float*)d_in[14];
  const float* ln2_b       = (const float*)d_in[15];
  const float* fc1_w       = (const float*)d_in[16];
  const float* fc1_b       = (const float*)d_in[17];
  const float* fc2_w       = (const float*)d_in[18];
  const float* fc2_b       = (const float*)d_in[19];

  // ---- workspace carve-up ----
  char* ws = (char*)d_ws;
  size_t off = 0;
  auto alloc = [&](size_t bytes)->void*{
    void* p = ws + off;
    off = (off + bytes + 255) & ~((size_t)255);
    return p;
  };
  float*    h_buf   = (float*)   alloc((size_t)LSEQ*DMODEL*4);        // running h
  float*    resid   = (float*)   alloc((size_t)LSEQ*DMODEL*4);
  float*    hout    = (float*)   alloc((size_t)LSEQ*DMODEL*4);
  _Float16* hn16    = (_Float16*)alloc((size_t)LSEQ*DMODEL*2);
  _Float16* x16     = (_Float16*)alloc((size_t)LSEQ*DMODEL*2);        // ln2 out
  _Float16* wbuf16  = (_Float16*)alloc((size_t)HMLP*DMODEL*2);        // largest weight in f16
  float*    xz      = (float*)   alloc((size_t)LSEQ*2*DINNER*4);
  float*    u_buf   = (float*)   alloc((size_t)LSEQ*DINNER*4);
  _Float16* u16     = (_Float16*)alloc((size_t)LSEQ*DINNER*2);
  float*    dbc     = (float*)   alloc((size_t)LSEQ*(RRANK+2*NSTATE)*4);
  _Float16* dbc16   = (_Float16*)alloc((size_t)LSEQ*(RRANK+2*NSTATE)*2);
  float*    delta   = (float*)   alloc((size_t)LSEQ*DINNER*4);
  float*    y_buf   = (float*)   alloc((size_t)LSEQ*DINNER*4);
  _Float16* yg16    = (_Float16*)alloc((size_t)LSEQ*DINNER*2);
  float*    t1      = (float*)   alloc((size_t)LSEQ*HMLP*4);          // fc1 out (post-gelu)
  _Float16* act16   = (_Float16*)alloc((size_t)LSEQ*HMLP*2);

  auto cvt = [&](const float* src, _Float16* dst, int n){
    cvt_f16_kernel<<<(n+255)/256, 256, 0, stream>>>(src, dst, n);
  };
  // blocks: 8 tile-rows per block, one tile-col per block
  auto gblocks = [](int M, int N){ return (M/128)*(N/16); };
  auto gshmem  = [](int Kdim){ return (size_t)16*Kdim*sizeof(_Float16); };

  // h = LN1(hidden)
  ln_kernel<<<LSEQ, 256, 0, stream>>>(hidden, ln1_w, ln1_b, h_buf, DMODEL);

  for (int i = 0; i < NLAYER; ++i){
    const float* in_w  = in_proj_w  + (size_t)i*2*DINNER*DMODEL;
    const float* cw    = conv_w     + (size_t)i*DINNER*KCONV;
    const float* cb    = conv_b     + (size_t)i*DINNER;
    const float* xw    = x_proj_w   + (size_t)i*(RRANK+2*NSTATE)*DINNER;
    const float* dtw   = dt_proj_w  + (size_t)i*DINNER*RRANK;
    const float* dtb   = dt_proj_b  + (size_t)i*DINNER;
    const float* Al    = A_log      + (size_t)i*DINNER*NSTATE;
    const float* Dpi   = D_param    + (size_t)i*DINNER;
    const float* ow    = out_proj_w + (size_t)i*DMODEL*DINNER;
    const float* bnw   = block_norm_w + (size_t)i*DMODEL;

    // residual (+)= h; hn16 = rms(residual)*bnw
    residual_rms_kernel<<<LSEQ, 256, 0, stream>>>(h_buf, resid, bnw, hn16, DMODEL, i > 0);

    // xz = hn @ in_w^T   (M=2048, N=2048, K=512)
    cvt(in_w, wbuf16, 2*DINNER*DMODEL);
    wmma_gemm_kernel<0><<<gblocks(LSEQ, 2*DINNER), 256, gshmem(DMODEL), stream>>>(
        hn16, DMODEL, wbuf16, DMODEL, xz, 2*DINNER, nullptr, nullptr,
        LSEQ, 2*DINNER, DMODEL);

    // u = silu(conv(x) + cb)
    conv_silu_kernel<<<(LSEQ*DINNER+255)/256, 256, 0, stream>>>(xz, cw, cb, u_buf, u16);

    // dbc = u @ xw^T   (M=2048, N=64, K=1024)
    cvt(xw, wbuf16, (RRANK+2*NSTATE)*DINNER);
    wmma_gemm_kernel<0><<<gblocks(LSEQ, RRANK+2*NSTATE), 256, gshmem(DINNER), stream>>>(
        u16, DINNER, wbuf16, DINNER, dbc, RRANK+2*NSTATE, nullptr, nullptr,
        LSEQ, RRANK+2*NSTATE, DINNER);

    // delta = softplus(dt @ dtw^T + dtb)   (M=2048, N=1024, K=32; A has lda=64)
    cvt(dbc, dbc16, LSEQ*(RRANK+2*NSTATE));
    cvt(dtw, wbuf16, DINNER*RRANK);
    wmma_gemm_kernel<1><<<gblocks(LSEQ, DINNER), 256, gshmem(RRANK), stream>>>(
        dbc16, RRANK+2*NSTATE, wbuf16, RRANK, delta, DINNER, dtb, nullptr,
        LSEQ, DINNER, RRANK);

    // y = ssm_scan(delta, B, C, u, A, Dp)
    ssm_scan_kernel<<<DINNER/256, 256, 0, stream>>>(delta, dbc, u_buf, Al, Dpi, y_buf);

    // yg16 = y * silu(z)
    gate_kernel<<<(LSEQ*DINNER+255)/256, 256, 0, stream>>>(y_buf, xz, yg16);

    // h = yg @ ow^T   (M=2048, N=512, K=1024)
    cvt(ow, wbuf16, DMODEL*DINNER);
    wmma_gemm_kernel<0><<<gblocks(LSEQ, DMODEL), 256, gshmem(DINNER), stream>>>(
        yg16, DINNER, wbuf16, DINNER, h_buf, DMODEL, nullptr, nullptr,
        LSEQ, DMODEL, DINNER);
  }

  // hout = shortcut + rms(h + residual)*norm_f_w
  final_rms_add_kernel<<<LSEQ, 256, 0, stream>>>(h_buf, resid, norm_f_w, hidden, hout, DMODEL);

  // x16 = LN2(hout)
  ln_f16_kernel<<<LSEQ, 256, 0, stream>>>(hout, ln2_w, ln2_b, x16, DMODEL);

  // t1 = gelu(x @ fc1^T + b1)   (M=2048, N=2048, K=512)
  cvt(fc1_w, wbuf16, HMLP*DMODEL);
  wmma_gemm_kernel<2><<<gblocks(LSEQ, HMLP), 256, gshmem(DMODEL), stream>>>(
      x16, DMODEL, wbuf16, DMODEL, t1, HMLP, fc1_b, nullptr,
      LSEQ, HMLP, DMODEL);

  // out = hout + t1 @ fc2^T + b2   (M=2048, N=512, K=2048)
  cvt(t1, act16, LSEQ*HMLP);
  cvt(fc2_w, wbuf16, DMODEL*HMLP);
  wmma_gemm_kernel<3><<<gblocks(LSEQ, DMODEL), 256, gshmem(HMLP), stream>>>(
      act16, HMLP, wbuf16, HMLP, (float*)d_out, DMODEL, fc2_b, hout,
      LSEQ, DMODEL, HMLP);
}
